// ModulatedConv2d_970662609054
// MI455X (gfx1250) — compile-verified, matrix operand order checked by reading, not run on hardware
//
#include <hip/hip_runtime.h>

// ---------------- problem constants (match reference) ----------------
#define BB   8
#define CC   512
#define OO   512
#define KK   3
#define WDIM 512
#define HH   64
#define WW   64
#define HP   66   /* padded height  */
#define WP   66   /* padded width   */

#define CONV_SCALE 0.014731391274719739f  /* 1/sqrt(512*9) */
#define MOD_SCALE  0.044194173824159216f  /* 1/sqrt(512)   */

typedef __attribute__((ext_vector_type(16))) __bf16        v16bf;
typedef __attribute__((ext_vector_type(8)))  float         v8f;
typedef __attribute__((ext_vector_type(4)))  unsigned int  u32x4;

// round-to-nearest-even float -> bf16 bits
__device__ __forceinline__ unsigned short f2bf(float f) {
    unsigned int u = __float_as_uint(f);
    u += 0x7FFFu + ((u >> 16) & 1u);
    return (unsigned short)(u >> 16);
}

// ---------------- kernel 1: modulation s[b,c] ----------------
__global__ void k_modulate(const float* __restrict__ w,
                           const float* __restrict__ mw,
                           const float* __restrict__ mb,
                           float* __restrict__ s) {
    int i = blockIdx.x * blockDim.x + threadIdx.x;
    if (i >= BB * CC) return;
    int b = i / CC, c = i % CC;
    const float* wr = w  + b * WDIM;
    const float* mr = mw + c * WDIM;
    float acc = 0.f;
    #pragma unroll 4
    for (int j = 0; j < WDIM; ++j) acc += wr[j] * mr[j];
    s[i] = acc * MOD_SCALE + mb[c] + 1.0f;
}

// ---------------- kernel 2: wsq[o,c] = sum_k base^2 ----------------
__global__ void k_wsq(const float* __restrict__ base, float* __restrict__ wsq) {
    int i = blockIdx.x * blockDim.x + threadIdx.x;
    if (i >= OO * CC) return;
    const float* p = base + i * (KK * KK);
    float acc = 0.f;
    #pragma unroll
    for (int k = 0; k < KK * KK; ++k) acc += p[k] * p[k];
    wsq[i] = acc;
}

// ---------------- kernel 3: demod d[b,o] ----------------
__global__ void k_demod(const float* __restrict__ s,
                        const float* __restrict__ wsq,
                        float* __restrict__ dm) {
    int i = blockIdx.x * blockDim.x + threadIdx.x;
    if (i >= BB * OO) return;
    int b = i / OO, o = i % OO;
    const float* sr = s   + b * CC;
    const float* wr = wsq + o * CC;
    float acc = 0.f;
    #pragma unroll 4
    for (int c = 0; c < CC; ++c) { float t = sr[c]; acc += t * t * wr[c]; }
    dm[i] = rsqrtf(acc * (CONV_SCALE * CONV_SCALE));
}

// ------- kernel 4: wmod[b][kk][o][c] = bf16(CONV_SCALE*base*s*d) -------
__global__ void k_wmod(const float* __restrict__ base,
                       const float* __restrict__ s,
                       const float* __restrict__ dm,
                       unsigned short* __restrict__ wmod) {
    int i = blockIdx.x * blockDim.x + threadIdx.x;  // B*9*O*C = 18,874,368
    if (i >= BB * 9 * OO * CC) return;
    int c  = i % CC;
    int o  = (i / CC) % OO;
    int kk = (i / (CC * OO)) % 9;
    int b  = i / (CC * OO * 9);
    float v = CONV_SCALE * base[(o * CC + c) * 9 + kk] * s[b * CC + c] * dm[b * OO + o];
    wmod[i] = f2bf(v);
}

// ------- kernel 5: padded transpose xT[b][h+1][w+1][c] = bf16(x[b][c][h][w]) -------
// Halo ring is written as zero, so the conv kernel needs no boundary predicate.
__global__ void k_xt(const float* __restrict__ x, unsigned short* __restrict__ xT) {
    int i = blockIdx.x * blockDim.x + threadIdx.x;  // B*HP*WP*C = 17,842,176
    if (i >= BB * HP * WP * CC) return;
    int c = i % CC;
    int w = (i / CC) % WP;
    int h = (i / (CC * WP)) % HP;
    int b = i / (CC * WP * HP);
    int hs = h - 1, ws = w - 1;
    unsigned short v = 0;
    if (hs >= 0 && hs < HH && ws >= 0 && ws < WW)
        v = f2bf(x[((b * CC + c) * HH + hs) * WW + ws]);
    xT[i] = v;
}

// ---------------- kernel 6: implicit-GEMM conv via WMMA ----------------
// grid = (32 row-pairs, 8 O-blocks, 8 batch), block = 128 (4 waves).
// Each wave: 64 output channels x 32 spatial positions (8 accumulators).
// Waves differ only in spatial position; A-fragment (weight) addresses are
// identical across all 4 waves -> deduped in WGP$.
// Per 32-deep K chunk: 12 b128 loads feed 8 WMMAs (1.5 loads/WMMA).
union Frag { u32x4 u[2]; v16bf v; };

#define WMMA_BF16(A, B, C) \
    __builtin_amdgcn_wmma_f32_16x16x32_bf16(false, (A).v, false, (B).v, (short)0, (C), false, false)

__global__ void __launch_bounds__(128)
k_conv(const unsigned short* __restrict__ xT,
       const unsigned short* __restrict__ wmod,
       float* __restrict__ out) {
    const int lane = threadIdx.x & 31;
    const int wave = threadIdx.x >> 5;
    const int n    = lane & 15;   // N column within 16x16 tile
    const int half = lane >> 4;   // k-half per ISA A/B layout

    const int b  = blockIdx.z;
    const int o0 = blockIdx.y * 64;              // 4 M-tiles of 16 (shared by waves)
    const int h  = blockIdx.x * 2 + (wave >> 1); // 2 rows per block
    const int w0 = (wave & 1) * 32;              // 2 N-tiles of 16 per wave

    v8f a00 = (v8f)0.0f, a10 = (v8f)0.0f, a20 = (v8f)0.0f, a30 = (v8f)0.0f;
    v8f a01 = (v8f)0.0f, a11 = (v8f)0.0f, a21 = (v8f)0.0f, a31 = (v8f)0.0f;

    #pragma unroll 1
    for (int kk = 0; kk < 9; ++kk) {
        const int kh = kk / 3, kw = kk % 3;
        // padded coords: hh = h+kh in [0,66), ww = w0+n+kw in [0,66)
        const unsigned short* xp =
            xT + ((b * HP + h + kh) * WP + (w0 + n + kw)) * CC + half * 8;
        const unsigned short* wp =
            wmod + ((b * 9 + kk) * OO + o0 + n) * CC + half * 8;

        #pragma unroll 4
        for (int c0 = 0; c0 < CC; c0 += 32) {
            // ---- issue ALL 12 b128 loads first (distinct register sets) ----
            const u32x4* pb0 = (const u32x4*)(xp + c0);            // spatial n
            const u32x4* pb1 = (const u32x4*)(xp + c0 + 16 * CC);  // spatial n+16
            const u32x4* q0  = (const u32x4*)(wp + c0);
            const u32x4* q1  = (const u32x4*)(wp + c0 + 16 * CC);
            const u32x4* q2  = (const u32x4*)(wp + c0 + 32 * CC);
            const u32x4* q3  = (const u32x4*)(wp + c0 + 48 * CC);

            Frag fb0, fb1, fa0, fa1, fa2, fa3;
            fb0.u[0] = pb0[0]; fb0.u[1] = pb0[2];
            fb1.u[0] = pb1[0]; fb1.u[1] = pb1[2];
            fa0.u[0] = q0[0];  fa0.u[1] = q0[2];
            fa1.u[0] = q1[0];  fa1.u[1] = q1[2];
            fa2.u[0] = q2[0];  fa2.u[1] = q2[2];
            fa3.u[0] = q3[0];  fa3.u[1] = q3[2];

            // ---- then 8 WMMAs (each A frag x2, each B frag x4) ----
            a00 = WMMA_BF16(fa0, fb0, a00);
            a10 = WMMA_BF16(fa1, fb0, a10);
            a20 = WMMA_BF16(fa2, fb0, a20);
            a30 = WMMA_BF16(fa3, fb0, a30);
            a01 = WMMA_BF16(fa0, fb1, a01);
            a11 = WMMA_BF16(fa1, fb1, a11);
            a21 = WMMA_BF16(fa2, fb1, a21);
            a31 = WMMA_BF16(fa3, fb1, a31);
        }
    }

    // ---- store D: VGPR r holds (M = r + 8*half, N = lane&15) ----
    const int base_hw = h * WW + w0 + n;
    #pragma unroll
    for (int r = 0; r < 8; ++r) {
        const int m = r + half * 8;
        out[((b * OO + (o0 +  0 + m)) * (HH * WW)) + base_hw]      = a00[r];
        out[((b * OO + (o0 + 16 + m)) * (HH * WW)) + base_hw]      = a10[r];
        out[((b * OO + (o0 + 32 + m)) * (HH * WW)) + base_hw]      = a20[r];
        out[((b * OO + (o0 + 48 + m)) * (HH * WW)) + base_hw]      = a30[r];
        out[((b * OO + (o0 +  0 + m)) * (HH * WW)) + base_hw + 16] = a01[r];
        out[((b * OO + (o0 + 16 + m)) * (HH * WW)) + base_hw + 16] = a11[r];
        out[((b * OO + (o0 + 32 + m)) * (HH * WW)) + base_hw + 16] = a21[r];
        out[((b * OO + (o0 + 48 + m)) * (HH * WW)) + base_hw + 16] = a31[r];
    }
}

// ---------------- host-side launcher ----------------
// Workspace layout (bytes):
//   s    : B*C*4        =     16,384   @ 0
//   wsq  : O*C*4        =  1,048,576   @ 16,384
//   dm   : B*O*4        =     16,384   @ 1,064,960
//   xT   : B*HP*WP*C*2  = 35,684,352   @ 1,081,344   (zero-padded halo)
//   wmod : B*9*O*C*2    = 37,748,736   @ 36,765,696
//   total ~74.5 MB
extern "C" void kernel_launch(void* const* d_in, const int* in_sizes, int n_in,
                              void* d_out, int out_size, void* d_ws, size_t ws_size,
                              hipStream_t stream) {
    const float* x    = (const float*)d_in[0];
    const float* w    = (const float*)d_in[1];
    const float* base = (const float*)d_in[2];
    const float* mw   = (const float*)d_in[3];
    const float* mb   = (const float*)d_in[4];
    float*       out  = (float*)d_out;

    char* ws = (char*)d_ws;
    float*          s    = (float*)(ws + 0);
    float*          wsq  = (float*)(ws + 16384);
    float*          dm   = (float*)(ws + 1064960);
    unsigned short* xT   = (unsigned short*)(ws + 1081344);
    unsigned short* wmod = (unsigned short*)(ws + 36765696);

    k_modulate<<<(BB * CC + 255) / 256, 256, 0, stream>>>(w, mw, mb, s);
    k_wsq<<<(OO * CC + 255) / 256, 256, 0, stream>>>(base, wsq);
    k_demod<<<(BB * OO + 255) / 256, 256, 0, stream>>>(s, wsq, dm);
    k_wmod<<<(BB * 9 * OO * CC + 255) / 256, 256, 0, stream>>>(base, s, dm, wmod);
    k_xt<<<(BB * HP * WP * CC + 255) / 256, 256, 0, stream>>>(x, xT);

    dim3 grid(32, 8, BB);
    k_conv<<<grid, 128, 0, stream>>>(xT, wmod, out);
}
